// EncoderRNN_24300924961384
// MI455X (gfx1250) — matrix operand order repaired and checked by reading, not compiled, not run on hardware
//
#include <hip/hip_runtime.h>
#include <hip/hip_bf16.h>
#include <stdint.h>

// Problem constants (V=32000, H=1024, T=512, batch=1)
#define H      1024
#define T      512
#define GATES  4
#define W_GATE (2 * H * H)   // per-gate [2H, H] weight block

typedef __attribute__((ext_vector_type(16))) __bf16 v16bf;
typedef __attribute__((ext_vector_type(8)))  float  v8f;

__device__ __forceinline__ unsigned short f32_bf16_rne(float f) {
    unsigned u = __float_as_uint(f);
    unsigned r = 0x7FFFu + ((u >> 16) & 1u);   // round-to-nearest-even
    u += r;
    return (unsigned short)(u >> 16);
}

__device__ __forceinline__ float sigf(float x) { return 1.0f / (1.0f + expf(-x)); }

// ---------------------------------------------------------------------------
// Kernel 1: embedding gather + bias, rounded to bf16 (feeds the x-side GEMM)
// emb_bf[t][k] = bf16(E[tok[t]][k] + b_e[k]), row-major [T][H]
// ---------------------------------------------------------------------------
__global__ void k_embed(const int* __restrict__ tok, const float* __restrict__ E,
                        const float* __restrict__ be, unsigned short* __restrict__ emb)
{
    int idx = blockIdx.x * blockDim.x + threadIdx.x;   // T*H threads
    int t = idx >> 10;
    int h = idx & (H - 1);
    float v = E[(size_t)tok[t] * H + h] + be[h];
    emb[idx] = f32_bf16_rne(v);
}

// ---------------------------------------------------------------------------
// Kernel 2: pack the x-part of W (rows 0..H-1 of the 2H input dim) into this code's
// wave32 bf16 B-fragment layout (B is 32x16 per WMMA: lane L holds column
// N=L%16, K = 16*(L/16)+j, j=0..15 contiguous per lane -> one 32B load/lane).
// Bpack layout: [dg=8][kt=32][nt=64][lane=32][16 bf16]
// ---------------------------------------------------------------------------
__global__ void k_packW(const float* __restrict__ Wr, const float* __restrict__ Wl,
                        unsigned short* __restrict__ Bp)
{
    int idx = blockIdx.x * blockDim.x + threadIdx.x;   // 8*32*64*32*16 = 8388608
    int j  = idx & 15;
    int ln = (idx >> 4) & 31;
    int nt = (idx >> 9) & 63;
    int kt = (idx >> 15) & 31;
    int dg = idx >> 20;                 // 0..7 : d*4+g
    int d = dg >> 2, g = dg & 3;
    int K = kt * 32 + ((ln >> 4) << 4) + j;
    int N = nt * 16 + (ln & 15);
    const float* W = d ? Wl : Wr;
    float v = W[(size_t)g * W_GATE + (size_t)K * H + N];
    Bp[idx] = f32_bf16_rne(v);
}

// ---------------------------------------------------------------------------
// Kernel 3: Gx[d][t][g*H+o] = b[d][g][o] + sum_k emb[t][k] * Wx[d][g][k][o]
// One 16x16 f32 tile per wave via v_wmma_f32_16x16x32_bf16, K-loop of 32.
// A fragment (16x32 bf16): lane L holds row M=L%16, K runs {b..b+15},
// b = 16*(L/16)  -> two contiguous 16B loads from row-major emb_bf.
// ---------------------------------------------------------------------------
__global__ void k_gates_x(const unsigned short* __restrict__ emb,
                          const unsigned short* __restrict__ Bp,
                          const float* __restrict__ br, const float* __restrict__ bl,
                          float* __restrict__ Gx)
{
    const int lane = threadIdx.x & 31;
    const int wave = threadIdx.x >> 5;
    const int flat = blockIdx.x * 8 + wave;   // 16384 tiles total
    const int nt = flat & 63;
    const int mt = (flat >> 6) & 31;
    const int dg = flat >> 11;                // 0..7
    const int d = dg >> 2, g = dg & 3;

    v8f acc = {0.f, 0.f, 0.f, 0.f, 0.f, 0.f, 0.f, 0.f};

    const int row = mt * 16 + (lane & 15);
    const unsigned short* arow = emb + (size_t)row * H;
    const int ksub = (lane >> 4) << 4;        // 0 or 16

    // base of this wave's B fragments in uint4 (16B) units; +4096 uint4 per kt
    const uint4* bptr0 = reinterpret_cast<const uint4*>(Bp) +
                         ((((size_t)dg * 2048 + nt) * 32 + lane) * 2);

    for (int kt = 0; kt < 32; ++kt) {
        union { v16bf v; uint4 q[2]; } A, B;
        const uint4* ap = reinterpret_cast<const uint4*>(arow + kt * 32 + ksub);
        A.q[0] = ap[0];          // K = klo .. klo+7
        A.q[1] = ap[1];          // K = klo+8 .. klo+15
        const uint4* bp = bptr0 + (size_t)kt * 4096;
        B.q[0] = bp[0];
        B.q[1] = bp[1];
        acc = __builtin_amdgcn_wmma_f32_16x16x32_bf16(
            /*neg_a=*/false, A.v, /*neg_b=*/false, B.v,
            /*c_mod=*/(short)0, acc, /*reuse_a=*/false, /*reuse_b=*/false);
    }

    const float* bias = d ? bl : br;
    const int o = nt * 16 + (lane & 15);
    const float bv = bias[g * H + o];
    const int mtop = (lane >> 4) << 3;
    #pragma unroll
    for (int v = 0; v < 8; ++v) {
        int t = mt * 16 + mtop + v;   // C/D layout: M = v + 8*(lane/16), N = lane%16
        Gx[(((size_t)d * T + t) * (GATES * H)) + g * H + o] = acc[v] + bv;
    }
}

// ---------------------------------------------------------------------------
// Kernel 4: persistent cooperative LSTM recurrence (both directions at once).
// 64 workgroups x 256 threads per direction; per step:
//   gates[j] = Gx[t][j] + sum_k h[k]*Wh[j..]   (coalesced column-block GEMV)
//   grid barrier; elementwise c/h update + annotation store; grid barrier.
// ---------------------------------------------------------------------------
#define NWG 64

__device__ __forceinline__ void grid_barrier(unsigned* cnt, unsigned* gen, unsigned nblk)
{
    __syncthreads();
    if (threadIdx.x == 0) {
        __threadfence();
        volatile unsigned* vgen = gen;
        unsigned g = *vgen;
        if (atomicAdd(cnt, 1u) == nblk - 1) {
            atomicExch(cnt, 0u);
            __threadfence();
            atomicAdd(gen, 1u);
        } else {
            while (*vgen == g) { __builtin_amdgcn_s_sleep(1); }
        }
        __threadfence();
    }
    __syncthreads();
}

__global__ void k_recurrent(const float* __restrict__ Wr, const float* __restrict__ Wl,
                            const float* __restrict__ Gx,
                            float* __restrict__ h0, float* __restrict__ h1,
                            float* __restrict__ c, float* __restrict__ gates,
                            unsigned* __restrict__ bars, float* __restrict__ out)
{
    const int d   = blockIdx.x / NWG;      // 0 = forward, 1 = backward
    const int w   = blockIdx.x % NWG;
    const int tid = threadIdx.x;

    const float* W   = d ? Wl : Wr;
    const float* Gxd = Gx + (size_t)d * T * (GATES * H);
    float* gd  = gates + d * (GATES * H);
    float* hd0 = h0 + d * H;
    float* hd1 = h1 + d * H;
    float* cd  = c  + d * H;
    unsigned* cnt = bars + d * 8;
    unsigned* gen = bars + d * 8 + 1;

    const int ol = tid & 63;               // output within this WG's 64-slice
    const int sl = tid >> 6;               // k-slice 0..3 (256 k's each)
    const int j  = w * 64 + ol;            // global gate output 0..4095
    const int g  = j >> 10;
    const int o  = j & (H - 1);
    // Wh column pointer: W[g][H + k][o], k = sl*256 + kk
    const float* wcol = W + (size_t)g * W_GATE + (size_t)H * H
                          + (size_t)(sl * 256) * H + o;

    __shared__ float red[256];

    for (int s = 0; s < T; ++s) {
        const float* hp = (s & 1) ? hd1 : hd0;
        float*       hn = (s & 1) ? hd0 : hd1;
        const int trow = d ? (T - 1 - s) : s;   // backward consumes reversed emb

        float acc = 0.f;
        const float* hk = hp + sl * 256;
        #pragma unroll 8
        for (int k = 0; k < 256; ++k)
            acc = fmaf(hk[k], wcol[(size_t)k * H], acc);

        red[tid] = acc;
        __syncthreads();
        if (tid < 64) {
            float v = red[tid] + red[tid + 64] + red[tid + 128] + red[tid + 192];
            gd[w * 64 + tid] = v + Gxd[(size_t)trow * (GATES * H) + w * 64 + tid];
        }
        __threadfence();
        grid_barrier(cnt, gen, NWG);

        const int gtid = w * 256 + tid;    // first 4 WGs handle the update
        if (gtid < H) {
            float ig = sigf(gd[gtid]);
            float og = sigf(gd[H + gtid]);
            float fg = sigf(gd[2 * H + gtid]);
            float cg = tanhf(gd[3 * H + gtid]);
            float cn = fg * cd[gtid] + ig * cg;
            float hh = og * tanhf(cn);
            cd[gtid] = cn;
            hn[gtid] = hh;
            // annotations[s][d*H + gtid]  (backward kept in reverse-iter order)
            out[(size_t)s * (2 * H) + d * H + gtid] = hh;
            if (d == 0 && s == T - 1) {
                out[(size_t)T * (2 * H) + gtid]     = hh;   // h_f
                out[(size_t)T * (2 * H) + H + gtid] = cn;   // c_f
            }
        }
        __threadfence();
        grid_barrier(cnt, gen, NWG);
    }
}

// ---------------------------------------------------------------------------
// Host launch
// ---------------------------------------------------------------------------
extern "C" void kernel_launch(void* const* d_in, const int* in_sizes, int n_in,
                              void* d_out, int out_size, void* d_ws, size_t ws_size,
                              hipStream_t stream)
{
    (void)in_sizes; (void)n_in; (void)out_size; (void)ws_size;
    const int*   tok = (const int*)  d_in[0];
    const float* E   = (const float*)d_in[1];
    const float* be  = (const float*)d_in[2];
    const float* Wr  = (const float*)d_in[3];
    const float* br  = (const float*)d_in[4];
    const float* Wl  = (const float*)d_in[5];
    const float* bl  = (const float*)d_in[6];
    float* out = (float*)d_out;

    char* ws = (char*)d_ws;
    unsigned short* emb_bf = (unsigned short*)ws;                        // 1 MB
    unsigned short* Bpack  = (unsigned short*)(ws + (1u << 20));         // 16 MB
    float* Gx              = (float*)(ws + (1u << 20) + (16u << 20));    // 16 MB
    char*  st = ws + (1u << 20) + (16u << 20) + (16u << 20);
    float* h0    = (float*)st;                    //  8 KB (2 dirs x H)
    float* h1    = (float*)(st + 8192);           //  8 KB
    float* c     = (float*)(st + 16384);          //  8 KB
    float* gates = (float*)(st + 24576);          // 32 KB (2 dirs x 4H)
    unsigned* bars = (unsigned*)(st + 24576 + 32768);
    const size_t st_bytes = 24576 + 32768 + 256;

    // zero h/c state, gate scratch and barrier counters (capture-safe)
    hipMemsetAsync(st, 0, st_bytes, stream);

    k_embed   <<<(T * H) / 256,      256, 0, stream>>>(tok, E, be, emb_bf);
    k_packW   <<<(8u << 20) / 256,   256, 0, stream>>>(Wr, Wl, Bpack);
    k_gates_x <<<2048,               256, 0, stream>>>(emb_bf, Bpack, br, bl, Gx);
    k_recurrent<<<2 * NWG,           256, 0, stream>>>(Wr, Wl, Gx, h0, h1, c,
                                                       gates, bars, out);
}
